// TransformerNet_38062000177197
// MI455X (gfx1250) — matrix operand order repaired and checked by reading, not compiled, not run on hardware
//
#include <hip/hip_runtime.h>
#include <hip/hip_bf16.h>
#include <math.h>

typedef __attribute__((ext_vector_type(16))) __bf16 v16bf;
typedef __attribute__((ext_vector_type(8)))  float  v8f;

// ---------------- device helpers ----------------

__device__ __forceinline__ unsigned short f2bf(float f) {
    unsigned u = __float_as_uint(f);
    return (unsigned short)((u + 0x7FFFu + ((u >> 16) & 1u)) >> 16); // RNE
}

// monotonic float<->uint encoding for atomicMax on signed floats
__device__ __forceinline__ unsigned fenc(float f) {
    unsigned u = __float_as_uint(f);
    return (u & 0x80000000u) ? ~u : (u | 0x80000000u);
}
__device__ __forceinline__ float fdec(unsigned e) {
    return (e & 0x80000000u) ? __uint_as_float(e ^ 0x80000000u)
                             : __uint_as_float(~e);
}

__device__ __forceinline__ float wave_reduce_sum(float v) {
#pragma unroll
    for (int off = 16; off > 0; off >>= 1) v += __shfl_xor(v, off, 32);
    return v;
}

// ---------------- elementwise / fill kernels ----------------

__global__ void f32_to_bf16_k(const float* __restrict__ in,
                              unsigned short* __restrict__ out, long n) {
    long i = (long)blockIdx.x * blockDim.x + threadIdx.x;
    if (i < n) out[i] = f2bf(in[i]);
}

__global__ void fill_f32_k(float* p, float v, long n) {
    long i = (long)blockIdx.x * blockDim.x + threadIdx.x;
    if (i < n) p[i] = v;
}

__global__ void fill_u32_k(unsigned* p, unsigned v, long n) {
    long i = (long)blockIdx.x * blockDim.x + threadIdx.x;
    if (i < n) p[i] = v;
}

// ---------------- WMMA bf16 GEMM:  D[N x M] = A[N x K] * W[K x M] + bias ----------------
// block tile 128x128, BK=32, 256 threads = 8 waves, wave tile 64x32 (4x2 wmma frags).
// Double-buffered LDS; A rows clamped (not guarded) so the hot loop is branch-free:
// out-of-range rows only feed D rows that the epilogue discards.

__global__ void gemm_bf16_wmma(const unsigned short* __restrict__ A,
                               const unsigned short* __restrict__ Wt,
                               const float* __restrict__ bias,
                               float* __restrict__ D,
                               int Nrows, int K, int M) {
    __shared__ unsigned short ldsA[2][128 * 40]; // [buf][row][k], pitch 40
    __shared__ unsigned short ldsB[2][128 * 40]; // [buf][col][k] (transposed)

    const int t    = threadIdx.x;
    const int lane = t & 31;
    const int l16  = lane & 15;
    const int hi   = (lane >> 4) & 1;
    const int wave = t >> 5;
    const int wm   = wave >> 2;  // 0..1  -> 64 rows each
    const int wn   = wave & 3;   // 0..3  -> 32 cols each
    const int rowBase = blockIdx.y * 128;
    const int colBase = blockIdx.x * 128;

    v8f acc[4][2];
#pragma unroll
    for (int i = 0; i < 4; ++i)
#pragma unroll
        for (int j = 0; j < 2; ++j)
#pragma unroll
            for (int e = 0; e < 8; ++e) acc[i][j][e] = 0.f;

    // stage one 128x32 A tile + 32x128 W tile (transposed) into LDS buffer `buf`
    auto load_tiles = [&](int k0, int buf) {
        // A: 4 threads/row, 2 row-passes, uint4 (8 bf16) per load, rows clamped
        const int ca = (t & 3) * 8;
#pragma unroll
        for (int it = 0; it < 2; ++it) {
            int r    = (t >> 2) + it * 64;
            int grow = rowBase + r;
            if (grow >= Nrows) grow = Nrows - 1; // clamp: dead rows discarded later
            uint4 val = *(const uint4*)(A + (size_t)grow * K + k0 + ca);
            *(uint4*)&ldsA[buf][r * 40 + ca] = val;
        }
        // W: 16 threads/k-row, 2 k-passes, store transposed [col][k]
        const int cb = (t & 15) * 8;
#pragma unroll
        for (int it = 0; it < 2; ++it) {
            int kk = (t >> 4) + it * 16;
            uint4 val = *(const uint4*)(Wt + (size_t)(k0 + kk) * M + colBase + cb);
            const unsigned short* s = (const unsigned short*)&val;
#pragma unroll
            for (int j = 0; j < 8; ++j)
                ldsB[buf][(cb + j) * 40 + kk] = s[j];
        }
    };

    const int nk = K >> 5;
    load_tiles(0, 0);

    for (int ki = 0; ki < nk; ++ki) {
        const int buf = ki & 1;
        __syncthreads();
        if (ki + 1 < nk) {
            load_tiles((ki + 1) << 5, buf ^ 1);
            if (ki + 2 < nk) {
                __builtin_prefetch(A + (size_t)(rowBase + (t >> 2)) * K + ((ki + 2) << 5), 0, 1);
                __builtin_prefetch(Wt + (size_t)(((ki + 2) << 5) + (t >> 4)) * M + colBase, 0, 1);
            }
        }

        union Frag { unsigned u[8]; v16bf v; };
        Frag fa[4], fb2[2];
        // A fragment (16-bit A 16x32 ISA layout): lanes0-15 rows; per-VGPR K below
#pragma unroll
        for (int fm = 0; fm < 4; ++fm) {
            int m = wm * 64 + fm * 16 + l16;
#pragma unroll
            for (int vv = 0; vv < 8; ++vv) {
                int kk = 2 * (vv & 3) + (hi ? 8 : 0) + ((vv >= 4) ? 16 : 0);
                fa[fm].u[vv] = *(const unsigned*)&ldsA[buf][m * 40 + kk];
            }
        }
        // B fragment: lane = column, VGPR v holds K = 2(v&3) + hi*8 + (v>=4)*16, +1
#pragma unroll
        for (int fn = 0; fn < 2; ++fn) {
            int n = wn * 32 + fn * 16 + l16;
#pragma unroll
            for (int vv = 0; vv < 8; ++vv) {
                int kk = 2 * (vv & 3) + (hi ? 8 : 0) + ((vv >= 4) ? 16 : 0);
                fb2[fn].u[vv] = *(const unsigned*)&ldsB[buf][n * 40 + kk];
            }
        }
#pragma unroll
        for (int fm = 0; fm < 4; ++fm)
#pragma unroll
            for (int fn = 0; fn < 2; ++fn)
                acc[fm][fn] = __builtin_amdgcn_wmma_f32_16x16x32_bf16(
                    false, fa[fm].v, false, fb2[fn].v, (short)0, acc[fm][fn],
                    false, false);
    }

    // epilogue: C/D layout — VGPR r: lanes0-15 M=r, lanes16-31 M=8+r; N=lane&15
#pragma unroll
    for (int fm = 0; fm < 4; ++fm) {
#pragma unroll
        for (int fn = 0; fn < 2; ++fn) {
            int col = colBase + wn * 32 + fn * 16 + l16;
            float b = bias ? bias[col] : 0.f;
#pragma unroll
            for (int r = 0; r < 8; ++r) {
                int row = rowBase + wm * 64 + fm * 16 + hi * 8 + r;
                if (row < Nrows)
                    D[(size_t)row * M + col] = acc[fm][fn][r] + b;
            }
        }
    }
}

// ---------------- edge attention kernels ----------------

// wave per (edge, head): alpha[e,h] = dot(q[dst], k[src]) * scale
__global__ void edge_logits_k(const float* __restrict__ q,
                              const float* __restrict__ k,
                              const int* __restrict__ src,
                              const int* __restrict__ dst,
                              float* __restrict__ alpha,
                              int E, int H, int C, float scale) {
    int w    = (blockIdx.x * blockDim.x + threadIdx.x) >> 5;
    int lane = threadIdx.x & 31;
    if (w >= E * H) return;
    int e = w / H, h = w % H;
    const float* qp = q + ((size_t)dst[e] * H + h) * C;
    const float* kp = k + ((size_t)src[e] * H + h) * C;
    float acc = 0.f;
    for (int c = lane; c < C; c += 32) acc += qp[c] * kp[c];
    acc = wave_reduce_sum(acc);
    if (lane == 0) alpha[w] = acc * scale;
}

__global__ void edge_seg_max_k(const float* __restrict__ alpha,
                               const int* __restrict__ dst,
                               unsigned* __restrict__ amax, int E, int H) {
    int w = blockIdx.x * blockDim.x + threadIdx.x;
    if (w >= E * H) return;
    int e = w / H, h = w % H;
    atomicMax(&amax[dst[e] * H + h], fenc(alpha[w]));
}

__global__ void edge_exp_sum_k(float* __restrict__ alpha,
                               const int* __restrict__ dst,
                               const unsigned* __restrict__ amax,
                               float* __restrict__ denom, int E, int H) {
    int w = blockIdx.x * blockDim.x + threadIdx.x;
    if (w >= E * H) return;
    int e = w / H, h = w % H;
    float m = fdec(amax[dst[e] * H + h]);
    if (!(m > -INFINITY && m < INFINITY)) m = 0.f;
    float ex = expf(alpha[w] - m);
    alpha[w] = ex;
    atomicAdd(&denom[dst[e] * H + h], ex);
}

// wave per (edge, head): out[dst] += (alpha/denom) * v[src]
__global__ void edge_scatter_k(const float* __restrict__ alpha,
                               const float* __restrict__ denom,
                               const float* __restrict__ v,
                               const int* __restrict__ src,
                               const int* __restrict__ dst,
                               float* __restrict__ out, int E, int H, int C) {
    int w    = (blockIdx.x * blockDim.x + threadIdx.x) >> 5;
    int lane = threadIdx.x & 31;
    if (w >= E * H) return;
    int e = w / H, h = w % H;
    int s = src[e], d = dst[e];
    float wgt = alpha[w] / (denom[d * H + h] + 1e-16f);
    const float* vp = v + ((size_t)s * H + h) * C;
    float* op = out + ((size_t)d * H + h) * C;
    for (int c = lane; c < C; c += 32) atomicAdd(&op[c], wgt * vp[c]);
}

// per-node: mean over heads + skip, ELU, emit f32 + bf16
__global__ void node_finalize_k(const float* __restrict__ att,
                                const float* __restrict__ skip,
                                float* __restrict__ hout,
                                unsigned short* __restrict__ hb,
                                int N, int H, int C) {
    long i = (long)blockIdx.x * blockDim.x + threadIdx.x;
    if (i >= (long)N * C) return;
    int n = (int)(i / C), c = (int)(i % C);
    float s = 0.f;
    for (int h = 0; h < H; ++h) s += att[((size_t)n * H + h) * C + c];
    float val = s / (float)H + skip[(size_t)n * C + c];
    val = val > 0.f ? val : (expf(val) - 1.f); // ELU
    hout[i] = val;
    hb[i]   = f2bf(val);
}

// ---------------- global attention pooling ----------------

__global__ void gate_logits_k(const float* __restrict__ h,
                              const float* __restrict__ Wg,
                              const float* __restrict__ bg,
                              float* __restrict__ gate, int N, int C) {
    int w    = (blockIdx.x * blockDim.x + threadIdx.x) >> 5;
    int lane = threadIdx.x & 31;
    if (w >= N) return;
    const float* hp = h + (size_t)w * C;
    float acc = 0.f;
    for (int c = lane; c < C; c += 32) acc += hp[c] * Wg[c];
    acc = wave_reduce_sum(acc);
    if (lane == 0) gate[w] = acc + bg[0];
}

__global__ void batch_max_k(const float* __restrict__ gate,
                            const int* __restrict__ batch,
                            unsigned* __restrict__ amax, int N) {
    int n = blockIdx.x * blockDim.x + threadIdx.x;
    if (n >= N) return;
    atomicMax(&amax[batch[n]], fenc(gate[n]));
}

__global__ void batch_exp_k(float* __restrict__ gate,
                            const int* __restrict__ batch,
                            const unsigned* __restrict__ amax,
                            float* __restrict__ denom, int N) {
    int n = blockIdx.x * blockDim.x + threadIdx.x;
    if (n >= N) return;
    float m = fdec(amax[batch[n]]);
    if (!(m > -INFINITY && m < INFINITY)) m = 0.f;
    float ex = expf(gate[n] - m);
    gate[n]  = ex;
    atomicAdd(&denom[batch[n]], ex);
}

__global__ void pool_scatter_k(const float* __restrict__ gate,
                               const float* __restrict__ denom,
                               const float* __restrict__ h,
                               const int* __restrict__ batch,
                               float* __restrict__ g, int N, int C) {
    long i = (long)blockIdx.x * blockDim.x + threadIdx.x;
    if (i >= (long)N * C) return;
    int n = (int)(i / C), c = (int)(i % C);
    int b = batch[n];
    float w = gate[n] / (denom[b] + 1e-16f);
    atomicAdd(&g[(size_t)b * C + c], w * h[i]);
}

__global__ void final_fc_k(const float* __restrict__ g,
                           const float* __restrict__ Wf,
                           const float* __restrict__ bf,
                           float* __restrict__ out, int B, int C, int OUT) {
    int i = blockIdx.x * blockDim.x + threadIdx.x;
    if (i >= B * OUT) return;
    int b = i / OUT, o = i % OUT;
    float acc = bf[o];
    for (int c = 0; c < C; ++c) acc += g[(size_t)b * C + c] * Wf[c * OUT + o];
    out[i] = acc;
}

// ---------------- host driver ----------------

extern "C" void kernel_launch(void* const* d_in, const int* in_sizes, int n_in,
                              void* d_out, int out_size, void* d_ws, size_t ws_size,
                              hipStream_t stream) {
    (void)in_sizes; (void)n_in; (void)out_size; (void)ws_size;
    const int N = 10000, E = 20000, F_IN = 128, H = 8, C1 = 512, C2 = 256,
              B = 64, OUT = 10;
    const unsigned ENC_NEG_INF = 0x007FFFFFu; // fenc(-inf)

    const float* x    = (const float*)d_in[0];
    const int*   ei   = (const int*)d_in[1];
    const int*   bat  = (const int*)d_in[2];
    const float* Wq1  = (const float*)d_in[3];  const float* bq1 = (const float*)d_in[4];
    const float* Wk1  = (const float*)d_in[5];  const float* bk1 = (const float*)d_in[6];
    const float* Wv1  = (const float*)d_in[7];  const float* bv1 = (const float*)d_in[8];
    const float* Ws1  = (const float*)d_in[9];  const float* bs1 = (const float*)d_in[10];
    const float* Wq2  = (const float*)d_in[11]; const float* bq2 = (const float*)d_in[12];
    const float* Wk2  = (const float*)d_in[13]; const float* bk2 = (const float*)d_in[14];
    const float* Wv2  = (const float*)d_in[15]; const float* bv2 = (const float*)d_in[16];
    const float* Ws2  = (const float*)d_in[17]; const float* bs2 = (const float*)d_in[18];
    const float* Wg   = (const float*)d_in[19]; const float* bg  = (const float*)d_in[20];
    const float* Wf   = (const float*)d_in[21]; const float* bfv = (const float*)d_in[22];
    const int* src = ei;
    const int* dst = ei + E;
    float* out = (float*)d_out;

    // workspace carve-out
    char* basep = (char*)d_ws;
    size_t off = 0;
    auto alloc = [&](size_t bytes) -> void* {
        void* p = basep + off;
        off = (off + bytes + 255) & ~(size_t)255;
        return p;
    };
    unsigned short* xb   = (unsigned short*)alloc((size_t)N * F_IN * 2);
    unsigned short* wq1b = (unsigned short*)alloc((size_t)F_IN * H * C1 * 2);
    unsigned short* wk1b = (unsigned short*)alloc((size_t)F_IN * H * C1 * 2);
    unsigned short* wv1b = (unsigned short*)alloc((size_t)F_IN * H * C1 * 2);
    unsigned short* ws1b = (unsigned short*)alloc((size_t)F_IN * C1 * 2);
    unsigned short* wq2b = (unsigned short*)alloc((size_t)C1 * H * C2 * 2);
    unsigned short* wk2b = (unsigned short*)alloc((size_t)C1 * H * C2 * 2);
    unsigned short* wv2b = (unsigned short*)alloc((size_t)C1 * H * C2 * 2);
    unsigned short* ws2b = (unsigned short*)alloc((size_t)C1 * C2 * 2);
    float* qb   = (float*)alloc((size_t)N * H * C1 * 4);  // reused for layer 2
    float* kb   = (float*)alloc((size_t)N * H * C1 * 4);
    float* vb   = (float*)alloc((size_t)N * H * C1 * 4);
    float* att  = (float*)alloc((size_t)N * H * C1 * 4);
    float* skip = (float*)alloc((size_t)N * C1 * 4);
    float* h1   = (float*)alloc((size_t)N * C1 * 4);
    unsigned short* h1b = (unsigned short*)alloc((size_t)N * C1 * 2);
    float* h2   = (float*)alloc((size_t)N * C2 * 4);
    float* alpha  = (float*)alloc((size_t)E * H * 4);
    unsigned* amax = (unsigned*)alloc((size_t)N * H * 4);
    float* denom  = (float*)alloc((size_t)N * H * 4);
    float* gate   = (float*)alloc((size_t)N * 4);
    unsigned* amaxB = (unsigned*)alloc((size_t)B * 4);
    float* denomB   = (float*)alloc((size_t)B * 4);
    float* gpool    = (float*)alloc((size_t)B * C2 * 4);

    auto blk = [](long n) { return (unsigned)((n + 255) / 256); };
    auto cvt = [&](const float* in, unsigned short* o, long n) {
        f32_to_bf16_k<<<blk(n), 256, 0, stream>>>(in, o, n);
    };
    auto fillf = [&](float* p, float v, long n) {
        fill_f32_k<<<blk(n), 256, 0, stream>>>(p, v, n);
    };
    auto fillu = [&](unsigned* p, unsigned v, long n) {
        fill_u32_k<<<blk(n), 256, 0, stream>>>(p, v, n);
    };
    auto gemm = [&](const unsigned short* A, const unsigned short* Wt,
                    const float* bias, float* D, int Nr, int K, int M) {
        dim3 g(M / 128, (Nr + 127) / 128);
        gemm_bf16_wmma<<<g, 256, 0, stream>>>(A, Wt, bias, D, Nr, K, M);
    };
    auto edge_phase = [&](const float* q, const float* k, const float* v,
                          float* o, int C) {
        fillu(amax, ENC_NEG_INF, (long)N * H);
        fillf(denom, 0.f, (long)N * H);
        fillf(o, 0.f, (long)N * H * C);
        long waves = (long)E * H;
        edge_logits_k<<<blk(waves * 32), 256, 0, stream>>>(q, k, src, dst, alpha,
                                                           E, H, C, rsqrtf((float)C));
        edge_seg_max_k<<<blk(waves), 256, 0, stream>>>(alpha, dst, amax, E, H);
        edge_exp_sum_k<<<blk(waves), 256, 0, stream>>>(alpha, dst, amax, denom, E, H);
        edge_scatter_k<<<blk(waves * 32), 256, 0, stream>>>(alpha, denom, v, src,
                                                            dst, o, E, H, C);
    };

    // ---- convert inputs to bf16 ----
    cvt(x,   xb,   (long)N * F_IN);
    cvt(Wq1, wq1b, (long)F_IN * H * C1);
    cvt(Wk1, wk1b, (long)F_IN * H * C1);
    cvt(Wv1, wv1b, (long)F_IN * H * C1);
    cvt(Ws1, ws1b, (long)F_IN * C1);
    cvt(Wq2, wq2b, (long)C1 * H * C2);
    cvt(Wk2, wk2b, (long)C1 * H * C2);
    cvt(Wv2, wv2b, (long)C1 * H * C2);
    cvt(Ws2, ws2b, (long)C1 * C2);

    // ---- layer 1 ----
    gemm(xb, wq1b, bq1, qb,   N, F_IN, H * C1);
    gemm(xb, wk1b, bk1, kb,   N, F_IN, H * C1);
    gemm(xb, wv1b, bv1, vb,   N, F_IN, H * C1);
    gemm(xb, ws1b, bs1, skip, N, F_IN, C1);
    edge_phase(qb, kb, vb, att, C1);
    node_finalize_k<<<blk((long)N * C1), 256, 0, stream>>>(att, skip, h1, h1b,
                                                           N, H, C1);

    // ---- layer 2 ----
    gemm(h1b, wq2b, bq2, qb,   N, C1, H * C2);
    gemm(h1b, wk2b, bk2, kb,   N, C1, H * C2);
    gemm(h1b, wv2b, bv2, vb,   N, C1, H * C2);
    gemm(h1b, ws2b, bs2, skip, N, C1, C2);
    edge_phase(qb, kb, vb, att, C2);
    node_finalize_k<<<blk((long)N * C2), 256, 0, stream>>>(att, skip, h2, h1b,
                                                           N, H, C2);

    // ---- global attention pooling ----
    gate_logits_k<<<blk((long)N * 32), 256, 0, stream>>>(h2, Wg, bg, gate, N, C2);
    fillu(amaxB, ENC_NEG_INF, B);
    fillf(denomB, 0.f, B);
    fillf(gpool, 0.f, (long)B * C2);
    batch_max_k<<<blk(N), 256, 0, stream>>>(gate, bat, amaxB, N);
    batch_exp_k<<<blk(N), 256, 0, stream>>>(gate, bat, amaxB, denomB, N);
    pool_scatter_k<<<blk((long)N * C2), 256, 0, stream>>>(gate, denomB, h2, bat,
                                                          gpool, N, C2);
    final_fc_k<<<blk(B * OUT), 256, 0, stream>>>(gpool, Wf, bfv, out, B, C2, OUT);
}
